// GNBlock_56427280335509
// MI455X (gfx1250) — matrix operand order repaired and checked by reading, not compiled, not run on hardware
//
#include <hip/hip_runtime.h>

#define NNODES 50000
#define NEDGES 800000
#define DNODE 64
#define DEDGE 64
#define HID 128

typedef __attribute__((ext_vector_type(16))) __bf16 v16bf;
typedef __attribute__((ext_vector_type(16))) unsigned short v16us;
typedef __attribute__((ext_vector_type(8))) float v8f;

// ---------- bf16 helpers ----------
__device__ __forceinline__ unsigned short bf16_rne(float f) {
    unsigned int u = __float_as_uint(f);
    u += 0x7FFFu + ((u >> 16) & 1u);
    return (unsigned short)(u >> 16);
}
__device__ __forceinline__ float bf16_to_f32(unsigned short h) {
    return __uint_as_float(((unsigned int)h) << 16);
}

// Cheap hi/lo split: hi = truncate-to-bf16(v) (exact residual), lo = truncate(v - hi).
// Residual error <= 2^-16 relative; compiler lowers the 16-bit packing to v_mov_b16 .l,.h moves.
__device__ __forceinline__ void split_trunc(float v, unsigned short& h, unsigned short& l) {
    unsigned int u = __float_as_uint(v);
    h = (unsigned short)(u >> 16);
    float lo = v - __uint_as_float(u & 0xFFFF0000u);  // exact
    l = (unsigned short)(__float_as_uint(lo) >> 16);
}

// Build a 16x32 bf16 A-fragment pair from fp32 row data.
// 16-bit A 16x32 layout: lane 0-15 -> elems 0..7 = K 0..7, elems 8..15 = K 16..23;
// lane 16-31 -> elems 0..7 = K 8..15, elems 8..15 = K 24..31.
__device__ __forceinline__ void make_a_frag(const float* __restrict__ src, int half,
                                            float scale, v16bf& ah, v16bf& al) {
    const int b0 = half * 8;
    float vals[16];
    {
        const float4 q0 = *(const float4*)(src + b0);
        const float4 q1 = *(const float4*)(src + b0 + 4);
        const float4 q2 = *(const float4*)(src + b0 + 16);
        const float4 q3 = *(const float4*)(src + b0 + 20);
        vals[0]=q0.x; vals[1]=q0.y; vals[2]=q0.z; vals[3]=q0.w;
        vals[4]=q1.x; vals[5]=q1.y; vals[6]=q1.z; vals[7]=q1.w;
        vals[8]=q2.x; vals[9]=q2.y; vals[10]=q2.z; vals[11]=q2.w;
        vals[12]=q3.x; vals[13]=q3.y; vals[14]=q3.z; vals[15]=q3.w;
    }
    v16us hu, lu;
#pragma unroll
    for (int j = 0; j < 16; ++j) {
        unsigned short h, l;
        split_trunc(vals[j] * scale, h, l);
        hu[j] = h;
        lu[j] = l;
    }
    ah = __builtin_bit_cast(v16bf, hu);
    al = __builtin_bit_cast(v16bf, lu);
}

// hi*hi + lo*hi + hi*lo  (drops lo*lo, ~fp32 accuracy with f32 accumulate)
__device__ __forceinline__ v8f wmma3v(v16bf ah, v16bf al, v16bf bh, v16bf bl, v8f c) {
    c = __builtin_amdgcn_wmma_f32_16x16x32_bf16(false, ah, false, bh, (short)0, c, false, false);
    c = __builtin_amdgcn_wmma_f32_16x16x32_bf16(false, al, false, bh, (short)0, c, false, false);
    c = __builtin_amdgcn_wmma_f32_16x16x32_bf16(false, ah, false, bl, (short)0, c, false, false);
    return c;
}

__device__ __forceinline__ void load_b(const unsigned short* __restrict__ wh,
                                       const unsigned short* __restrict__ wl,
                                       int frag, int lane, v16bf& bh, v16bf& bl) {
    bh = *(const v16bf*)(wh + ((size_t)frag * 32 + lane) * 16);
    bl = *(const v16bf*)(wl + ((size_t)frag * 32 + lane) * 16);
}

// ---------- prep kernels ----------
__global__ void zero_kernel(float* __restrict__ p, int n) {
    int t = blockIdx.x * blockDim.x + threadIdx.x;
    if (t < n) p[t] = 0.0f;
}

__global__ void count_kernel(const int* __restrict__ row, float* __restrict__ cnt) {
    int t = blockIdx.x * blockDim.x + threadIdx.x;
    if (t < NEDGES) atomicAdd(&cnt[row[t]], 1.0f);
}

__global__ void rdenom_kernel(const float* __restrict__ cnt, float* __restrict__ rd) {
    int t = blockIdx.x * blockDim.x + threadIdx.x;
    if (t < NNODES) rd[t] = 1.0f / fmaxf(cnt[t], 1.0f);
}

// Pack fp32 KxN weight into bf16 hi/lo B-fragments (RNE, one-time).
// B 32x16 layout: lane 0-15 -> N=nt*16+lane, elems 0..7 = K 0..7, 8..15 = K 16..23; lane 16-31 -> elems 0..7 = K 8..15, 8..15 = K 24..31.
__global__ void pack_w_kernel(const float* __restrict__ W, int K, int N,
                              unsigned short* __restrict__ hi, unsigned short* __restrict__ lo,
                              int combine) {
    int t = blockIdx.x * blockDim.x + threadIdx.x;
    if (t >= K * N) return;
    int k = t / N, n = t % N;
    float v;
    if (combine)
        v = (k < 64) ? (W[k * N + n] + W[(k + 64) * N + n]) : W[(k + 64) * N + n];
    else
        v = W[k * N + n];
    int kt = k >> 5, kr = k & 31, nt = n >> 4, nl = n & 15;
    int lane = nl + ((kr & 8) ? 16 : 0);
    int elem = (kr & 7) + ((kr & 16) ? 8 : 0);
    int ntiles = N >> 4;
    size_t off = (((size_t)(kt * ntiles + nt)) * 32 + lane) * 16 + elem;
    unsigned short h = bf16_rne(v);
    hi[off] = h;
    lo[off] = bf16_rne(v - bf16_to_f32(h));
}

// ---------- edge MLP: 32 edges (two 16-row M-tiles) per wave ----------
// e_new = relu([x[r]|ea] @ Wc1 + be1) @ We2 + be2 ; eout = ein + e_new ; agg += e_new (scatter)
__launch_bounds__(32)
__global__ void edge_kernel(const float* __restrict__ x, const float* __restrict__ ein,
                            const int* __restrict__ rowidx,
                            const unsigned short* __restrict__ w1h, const unsigned short* __restrict__ w1l,
                            const unsigned short* __restrict__ w2h, const unsigned short* __restrict__ w2l,
                            const float* __restrict__ be1, const float* __restrict__ be2,
                            float* __restrict__ eout, float* __restrict__ agg) {
    __shared__ float h_lds[32 * HID];  // 16 KB
    const int lane = threadIdx.x;
    const int half = lane >> 4;
    const int m = lane & 15;
    const long long e0 = (long long)blockIdx.x * 32;
    const int r0 = rowidx[e0 + m];
    const int r1 = rowidx[e0 + 16 + m];

    v8f acc1[2][8];
#pragma unroll
    for (int nt = 0; nt < 8; ++nt) {
        float b = be1[nt * 16 + m];
#pragma unroll
        for (int i = 0; i < 8; ++i) { acc1[0][nt][i] = b; acc1[1][nt][i] = b; }
    }

#pragma unroll
    for (int kt = 0; kt < 4; ++kt) {
        const float* s0 = (kt < 2) ? (x + (long long)r0 * DNODE + kt * 32)
                                   : (ein + (e0 + m) * DEDGE + (kt - 2) * 32);
        const float* s1 = (kt < 2) ? (x + (long long)r1 * DNODE + kt * 32)
                                   : (ein + (e0 + 16 + m) * DEDGE + (kt - 2) * 32);
        v16bf ah0, al0, ah1, al1;
        make_a_frag(s0, half, 1.0f, ah0, al0);
        make_a_frag(s1, half, 1.0f, ah1, al1);
#pragma unroll
        for (int nt = 0; nt < 8; ++nt) {
            v16bf bh, bl;
            load_b(w1h, w1l, kt * 8 + nt, lane, bh, bl);
            acc1[0][nt] = wmma3v(ah0, al0, bh, bl, acc1[0][nt]);
            acc1[1][nt] = wmma3v(ah1, al1, bh, bl, acc1[1][nt]);
        }
    }

    // ReLU + transpose D-layout -> A-layout via LDS (both tiles)
#pragma unroll
    for (int ti = 0; ti < 2; ++ti)
#pragma unroll
        for (int nt = 0; nt < 8; ++nt)
#pragma unroll
            for (int i = 0; i < 8; ++i)
                h_lds[(ti * 16 + i + 8 * half) * HID + nt * 16 + m] = fmaxf(acc1[ti][nt][i], 0.0f);
    __syncthreads();

    v8f acc2[2][4];
#pragma unroll
    for (int nt = 0; nt < 4; ++nt) {
        float b = be2[nt * 16 + m];
#pragma unroll
        for (int i = 0; i < 8; ++i) { acc2[0][nt][i] = b; acc2[1][nt][i] = b; }
    }
#pragma unroll
    for (int kt = 0; kt < 4; ++kt) {
        v16bf ah0, al0, ah1, al1;
        make_a_frag(&h_lds[m * HID + kt * 32], half, 1.0f, ah0, al0);
        make_a_frag(&h_lds[(16 + m) * HID + kt * 32], half, 1.0f, ah1, al1);
#pragma unroll
        for (int nt = 0; nt < 4; ++nt) {
            v16bf bh, bl;
            load_b(w2h, w2l, kt * 4 + nt, lane, bh, bl);
            acc2[0][nt] = wmma3v(ah0, al0, bh, bl, acc2[0][nt]);
            acc2[1][nt] = wmma3v(ah1, al1, bh, bl, acc2[1][nt]);
        }
    }

    // D element (vgpr i, lane) -> row M=i+8*half, col N=m (+16*nt)
#pragma unroll
    for (int ti = 0; ti < 2; ++ti)
#pragma unroll
        for (int i = 0; i < 8; ++i) {
            long long ge = e0 + ti * 16 + i + 8 * half;
            int rr = rowidx[ge];
#pragma unroll
            for (int nt = 0; nt < 4; ++nt) {
                int col = nt * 16 + m;
                float v = acc2[ti][nt][i];
                eout[ge * DEDGE + col] = ein[ge * DEDGE + col] + v;
                atomicAdd(agg + (long long)rr * DEDGE + col, v);
            }
        }
}

// ---------- node MLP: 16 nodes per wave ----------
__launch_bounds__(32)
__global__ void node_kernel(const float* __restrict__ x, const float* __restrict__ agg,
                            const float* __restrict__ rdenom,
                            const unsigned short* __restrict__ w1h, const unsigned short* __restrict__ w1l,
                            const unsigned short* __restrict__ w2h, const unsigned short* __restrict__ w2l,
                            const float* __restrict__ bn1, const float* __restrict__ bn2,
                            float* __restrict__ xout) {
    __shared__ float h_lds[16 * HID];
    const int lane = threadIdx.x;
    const int half = lane >> 4;
    const int m = lane & 15;
    const long long n0 = (long long)blockIdx.x * 16;
    const float scale = rdenom[n0 + m];

    v8f acc1[8];
#pragma unroll
    for (int nt = 0; nt < 8; ++nt) {
        float b = bn1[nt * 16 + m];
#pragma unroll
        for (int i = 0; i < 8; ++i) acc1[nt][i] = b;
    }

#pragma unroll
    for (int kt = 0; kt < 4; ++kt) {
        const float* src = (kt < 2) ? (x + (n0 + m) * DNODE + kt * 32)
                                    : (agg + (n0 + m) * DEDGE + (kt - 2) * 32);
        v16bf ah, al;
        make_a_frag(src, half, (kt < 2) ? 1.0f : scale, ah, al);
#pragma unroll
        for (int nt = 0; nt < 8; ++nt) {
            v16bf bh, bl;
            load_b(w1h, w1l, kt * 8 + nt, lane, bh, bl);
            acc1[nt] = wmma3v(ah, al, bh, bl, acc1[nt]);
        }
    }

#pragma unroll
    for (int nt = 0; nt < 8; ++nt)
#pragma unroll
        for (int i = 0; i < 8; ++i)
            h_lds[(i + 8 * half) * HID + nt * 16 + m] = fmaxf(acc1[nt][i], 0.0f);
    __syncthreads();

    v8f acc2[4];
#pragma unroll
    for (int nt = 0; nt < 4; ++nt) {
        float b = bn2[nt * 16 + m];
#pragma unroll
        for (int i = 0; i < 8; ++i) acc2[nt][i] = b;
    }
#pragma unroll
    for (int kt = 0; kt < 4; ++kt) {
        v16bf ah, al;
        make_a_frag(&h_lds[m * HID + kt * 32], half, 1.0f, ah, al);
#pragma unroll
        for (int nt = 0; nt < 4; ++nt) {
            v16bf bh, bl;
            load_b(w2h, w2l, kt * 4 + nt, lane, bh, bl);
            acc2[nt] = wmma3v(ah, al, bh, bl, acc2[nt]);
        }
    }

#pragma unroll
    for (int i = 0; i < 8; ++i) {
        long long gn = n0 + i + 8 * half;
#pragma unroll
        for (int nt = 0; nt < 4; ++nt) {
            int col = nt * 16 + m;
            xout[gn * DNODE + col] = x[gn * DNODE + col] + acc2[nt][i];
        }
    }
}

extern "C" void kernel_launch(void* const* d_in, const int* in_sizes, int n_in,
                              void* d_out, int out_size, void* d_ws, size_t ws_size,
                              hipStream_t stream) {
    (void)in_sizes; (void)n_in; (void)out_size; (void)ws_size;
    const float* x0   = (const float*)d_in[0];
    const float* ea0  = (const float*)d_in[1];
    const float* We1  = (const float*)d_in[2];
    const float* be1  = (const float*)d_in[3];
    const float* We2  = (const float*)d_in[4];
    const float* be2  = (const float*)d_in[5];
    const float* Wn1  = (const float*)d_in[6];
    const float* bn1  = (const float*)d_in[7];
    const float* Wn2  = (const float*)d_in[8];
    const float* bn2  = (const float*)d_in[9];
    const int*   row  = (const int*)d_in[10];  // edge_index[0]; col==row per reference

    float* out_x = (float*)d_out;                       // [NNODES, 64]
    float* out_e = out_x + (size_t)NNODES * DNODE;      // [NEDGES, 64]

    char* ws = (char*)d_ws;
    size_t off = 0;
    auto take = [&](size_t bytes) { size_t o = off; off = (off + bytes + 255) & ~(size_t)255; return o; };
    float* agg    = (float*)(ws + take((size_t)NNODES * DEDGE * 4));
    float* cnt    = (float*)(ws + take((size_t)NNODES * 4));
    float* rdenom = (float*)(ws + take((size_t)NNODES * 4));
    float* x1     = (float*)(ws + take((size_t)NNODES * DNODE * 4));
    float* e1     = (float*)(ws + take((size_t)NEDGES * DEDGE * 4));
    unsigned short* w1h  = (unsigned short*)(ws + take((size_t)128 * 128 * 2));
    unsigned short* w1l  = (unsigned short*)(ws + take((size_t)128 * 128 * 2));
    unsigned short* w2h  = (unsigned short*)(ws + take((size_t)128 * 64 * 2));
    unsigned short* w2l  = (unsigned short*)(ws + take((size_t)128 * 64 * 2));
    unsigned short* wn1h = (unsigned short*)(ws + take((size_t)128 * 128 * 2));
    unsigned short* wn1l = (unsigned short*)(ws + take((size_t)128 * 128 * 2));
    unsigned short* wn2h = (unsigned short*)(ws + take((size_t)128 * 64 * 2));
    unsigned short* wn2l = (unsigned short*)(ws + take((size_t)128 * 64 * 2));

    pack_w_kernel<<<(128 * 128 + 255) / 256, 256, 0, stream>>>(We1, 128, 128, w1h, w1l, 1);
    pack_w_kernel<<<(128 * 64 + 255) / 256, 256, 0, stream>>>(We2, 128, 64, w2h, w2l, 0);
    pack_w_kernel<<<(128 * 128 + 255) / 256, 256, 0, stream>>>(Wn1, 128, 128, wn1h, wn1l, 0);
    pack_w_kernel<<<(128 * 64 + 255) / 256, 256, 0, stream>>>(Wn2, 128, 64, wn2h, wn2l, 0);

    zero_kernel<<<(NNODES + 255) / 256, 256, 0, stream>>>(cnt, NNODES);
    count_kernel<<<(NEDGES + 255) / 256, 256, 0, stream>>>(row, cnt);
    rdenom_kernel<<<(NNODES + 255) / 256, 256, 0, stream>>>(cnt, rdenom);

    const int AGG_N = NNODES * DEDGE;

    // iteration 0: inputs -> ws buffers
    zero_kernel<<<(AGG_N + 255) / 256, 256, 0, stream>>>(agg, AGG_N);
    edge_kernel<<<NEDGES / 32, 32, 0, stream>>>(x0, ea0, row, w1h, w1l, w2h, w2l, be1, be2, e1, agg);
    node_kernel<<<NNODES / 16, 32, 0, stream>>>(x0, agg, rdenom, wn1h, wn1l, wn2h, wn2l, bn1, bn2, x1);

    // iteration 1: ws buffers -> d_out
    zero_kernel<<<(AGG_N + 255) / 256, 256, 0, stream>>>(agg, AGG_N);
    edge_kernel<<<NEDGES / 32, 32, 0, stream>>>(x1, e1, row, w1h, w1l, w2h, w2l, be1, be2, out_e, agg);
    node_kernel<<<NNODES / 16, 32, 0, stream>>>(x1, agg, rdenom, wn1h, wn1l, wn2h, wn2l, bn1, bn2, out_x);
}
